// SwitchLinear_53137335386911
// MI455X (gfx1250) — matrix operand order repaired and checked by reading, not compiled, not run on hardware
//
#include <hip/hip_runtime.h>
#include <math.h>

typedef __bf16 bf16;
typedef bf16 bf16x8 __attribute__((ext_vector_type(8)));
typedef bf16 v16bf __attribute__((ext_vector_type(16)));
typedef float v8f  __attribute__((ext_vector_type(8)));
typedef int  v4i  __attribute__((ext_vector_type(4)));

#define T_TOK 9216
#define D_IN  1024
#define DD    2048
#define EE    4
#define HH    8192
#define CAPC  3456
#define SLOTS (EE * CAPC)

__device__ __forceinline__ float gelu_erf(float x) {
    return 0.5f * x * (1.0f + erff(x * 0.70710678118654752f));
}

// ---------------- gfx1250 async global->LDS copy (bypasses VGPRs, ASYNCcnt) ----------------
#if defined(__has_builtin)
#  if __has_builtin(__builtin_amdgcn_global_load_async_to_lds_b128)
#    define HAVE_ASYNC_LDS_BUILTIN 1
#  endif
#  if __has_builtin(__builtin_amdgcn_s_wait_asynccnt)
#    define HAVE_WAIT_ASYNC_BUILTIN 1
#  endif
#endif

typedef __attribute__((address_space(1))) v4i* gptr_v4i;
typedef __attribute__((address_space(3))) v4i* lptr_v4i;

__device__ __forceinline__ void async_b128(lptr_v4i lds, const bf16* g) {
#ifdef HAVE_ASYNC_LDS_BUILTIN
    __builtin_amdgcn_global_load_async_to_lds_b128((gptr_v4i)g, lds, 0, 0);
#else
    unsigned off = (unsigned)(size_t)lds;
    asm volatile("global_load_async_to_lds_b128 %0, %1, off" :: "v"(off), "v"(g) : "memory");
#endif
}

template <int N>
__device__ __forceinline__ void wait_async() {
#ifdef HAVE_WAIT_ASYNC_BUILTIN
    __builtin_amdgcn_s_wait_asynccnt((short)N);
#else
    asm volatile("s_wait_asynccnt %0" :: "n"(N) : "memory");
#endif
}

// ---------------- transpose + f32 -> bf16 conversion ----------------
// src: [K, N] f32 row-major ; dst: [N, K] bf16 row-major ; blockIdx.z = batch
__global__ void transpose_to_bf16(const float* __restrict__ src, bf16* __restrict__ dst,
                                  int K, int N) {
    size_t zo = (size_t)blockIdx.z * (size_t)K * (size_t)N;
    src += zo; dst += zo;
    __shared__ float tile[32][33];
    int k0 = blockIdx.x * 32, n0 = blockIdx.y * 32;
    int tx = threadIdx.x, ty = threadIdx.y;
#pragma unroll
    for (int i = 0; i < 4; ++i)
        tile[ty + i * 8][tx] = src[(size_t)(k0 + ty + i * 8) * N + (n0 + tx)];
    __syncthreads();
#pragma unroll
    for (int i = 0; i < 4; ++i)
        dst[(size_t)(n0 + ty + i * 8) * K + (k0 + tx)] = (bf16)tile[tx][ty + i * 8];
}

// ---------------- LayerNorm -> bf16 ----------------
__global__ void ln_to_bf16(const float* __restrict__ x, const float* __restrict__ g,
                           const float* __restrict__ b, bf16* __restrict__ xn) {
    int tok = blockIdx.x;
    int tid = threadIdx.x;
    const float* xr = x + (size_t)tok * D_IN;
    float v[4];
    float s = 0.f, ss = 0.f;
#pragma unroll
    for (int i = 0; i < 4; ++i) {
        v[i] = xr[tid + i * 256];
        s += v[i]; ss += v[i] * v[i];
    }
    __shared__ float r0[256], r1[256];
    r0[tid] = s; r1[tid] = ss;
    __syncthreads();
    for (int off = 128; off > 0; off >>= 1) {
        if (tid < off) { r0[tid] += r0[tid + off]; r1[tid] += r1[tid + off]; }
        __syncthreads();
    }
    float mu  = r0[0] * (1.0f / D_IN);
    float var = r1[0] * (1.0f / D_IN) - mu * mu;
    float rstd = rsqrtf(var + 1e-5f);
    bf16* o = xn + (size_t)tok * D_IN;
#pragma unroll
    for (int i = 0; i < 4; ++i) {
        int c = tid + i * 256;
        o[c] = (bf16)((v[i] - mu) * rstd * g[c] + b[c]);
    }
}

// ---------------- WMMA bf16 GEMM: C = act(A[M,K] * Bt[N,K]^T + bias) ----------------
// 256 threads = 8 waves (4 M x 2 N), block tile 128x128, K-step 32.
// Double-buffered LDS fed by async global->LDS copies; loop unrolled x2 so the
// buffer parity (and all LDS destination pointers) are compile-time static.
// Requires K % 64 == 0 (holds: K = 1024 / 2048 / 8192).
__global__ __launch_bounds__(256) void gemm_wmma_bf16(
    const bf16* __restrict__ A, const bf16* __restrict__ Bt,
    const float* __restrict__ bias,
    float* __restrict__ outF, bf16* __restrict__ outB,
    int M, int N, int K,
    long sA_, long sB_, long sBias, long sOF, long sOB, int act)
{
    int z = blockIdx.z;
    A  += (size_t)z * sA_;
    Bt += (size_t)z * sB_;
    if (bias) bias += (size_t)z * sBias;
    if (outF) outF += (size_t)z * sOF;
    if (outB) outB += (size_t)z * sOB;

    // 40-elem (80B) row pitch: 16B aligned, bank-spread. 2 buffers each: 40KB total.
    __shared__ bf16 shA[2][128 * 40];
    __shared__ bf16 shB[2][128 * 40];

    int tid = threadIdx.x;
    int m0 = blockIdx.y * 128;
    int n0 = blockIdx.x * 128;

    int lr = tid >> 2;          // 0..63
    int lc = (tid & 3) * 8;     // 0,8,16,24
    const bf16* gA0 = A  + (size_t)(m0 + lr)      * K + lc;
    const bf16* gA1 = A  + (size_t)(m0 + lr + 64) * K + lc;
    const bf16* gB0 = Bt + (size_t)(n0 + lr)      * K + lc;
    const bf16* gB1 = Bt + (size_t)(n0 + lr + 64) * K + lc;

    // LDS destinations, addrspacecast hoisted out of the loop (no per-iter null checks)
    lptr_v4i dA0[2] = { (lptr_v4i)&shA[0][lr * 40 + lc],        (lptr_v4i)&shA[1][lr * 40 + lc] };
    lptr_v4i dA1[2] = { (lptr_v4i)&shA[0][(lr + 64) * 40 + lc], (lptr_v4i)&shA[1][(lr + 64) * 40 + lc] };
    lptr_v4i dB0[2] = { (lptr_v4i)&shB[0][lr * 40 + lc],        (lptr_v4i)&shB[1][lr * 40 + lc] };
    lptr_v4i dB1[2] = { (lptr_v4i)&shB[0][(lr + 64) * 40 + lc], (lptr_v4i)&shB[1][(lr + 64) * 40 + lc] };

    int lane = tid & 31;
    int wave = tid >> 5;
    int wm = wave >> 1;         // 0..3
    int wn = wave & 1;          // 0..1
    int half = lane >> 4;
    int l16  = lane & 15;

    v8f acc[2][4];
#pragma unroll
    for (int i = 0; i < 2; ++i)
#pragma unroll
        for (int j = 0; j < 4; ++j)
            acc[i][j] = v8f{0.f, 0.f, 0.f, 0.f, 0.f, 0.f, 0.f, 0.f};

    const int nk = K >> 5;      // K / 32 tiles (even)

    auto issue = [&](int pbuf, int kt) {
        int kk = kt << 5;
        async_b128(dA0[pbuf], gA0 + kk);
        async_b128(dA1[pbuf], gA1 + kk);
        async_b128(dB0[pbuf], gB0 + kk);
        async_b128(dB1[pbuf], gB1 + kk);
    };
    auto compute = [&](const bf16* sa, const bf16* sb) {
        v16bf afrag[2], bfrag[4];
#pragma unroll
        for (int mt = 0; mt < 2; ++mt) {
            // A 16x32 frag: lane holds K in [8h,8h+8) and [16+8h,16+8h+8) of row
            int row = wm * 32 + mt * 16 + l16;
            bf16x8 lo = *(const bf16x8*)&sa[row * 40 + half * 8];
            bf16x8 hi = *(const bf16x8*)&sa[row * 40 + 16 + half * 8];
            afrag[mt] = __builtin_shufflevector(lo, hi, 0,1,2,3,4,5,6,7,8,9,10,11,12,13,14,15);
        }
#pragma unroll
        for (int nt = 0; nt < 4; ++nt) {
            // B 32x16 frag: lane holds K in [8h,8h+8) and [16+8h,16+8h+8) of column (Bt row)
            int nr = wn * 64 + nt * 16 + l16;
            bf16x8 lo = *(const bf16x8*)&sb[nr * 40 + half * 8];
            bf16x8 hi = *(const bf16x8*)&sb[nr * 40 + 16 + half * 8];
            bfrag[nt] = __builtin_shufflevector(lo, hi, 0,1,2,3,4,5,6,7,8,9,10,11,12,13,14,15);
        }
#pragma unroll
        for (int mt = 0; mt < 2; ++mt)
#pragma unroll
            for (int nt = 0; nt < 4; ++nt)
                acc[mt][nt] = __builtin_amdgcn_wmma_f32_16x16x32_bf16(
                    false, afrag[mt], false, bfrag[nt], (short)0, acc[mt][nt], false, false);
    };

    // software pipeline, unrolled x2 (static buffer parity)
    issue(0, 0);
    for (int i = 0; i < nk; i += 2) {
        // ---- tile i (buffer 0); tile i+1 always exists (nk even) ----
        issue(1, i + 1);
        if (i + 2 < nk) {
            __builtin_prefetch(gA0 + ((i + 2) << 5), 0, 1);
            __builtin_prefetch(gB0 + ((i + 2) << 5), 0, 1);
        }
        wait_async<4>();        // in-order: oldest 4 (tile i) have landed in LDS
        __syncthreads();
        compute(shA[0], shB[0]);
        __syncthreads();

        // ---- tile i+1 (buffer 1) ----
        if (i + 2 < nk) {
            issue(0, i + 2);
            wait_async<4>();
        } else {
            wait_async<0>();
        }
        __syncthreads();
        compute(shA[1], shB[1]);
        __syncthreads();
    }

#pragma unroll
    for (int nt = 0; nt < 4; ++nt) {
        int col = n0 + wn * 64 + nt * 16 + l16;
        float bv = bias ? bias[col] : 0.0f;
#pragma unroll
        for (int mt = 0; mt < 2; ++mt) {
#pragma unroll
            for (int r = 0; r < 8; ++r) {
                int row = m0 + wm * 32 + mt * 16 + half * 8 + r;
                float vv = acc[mt][nt][r] + bv;
                if (act) vv = gelu_erf(vv);
                size_t o = (size_t)row * N + col;
                if (outF) outF[o] = vv;
                if (outB) outB[o] = (bf16)vv;
            }
        }
    }
}

// ---------------- Router: one wave32 per token ----------------
__global__ void router_kernel(const float* __restrict__ t, const float* __restrict__ swW,
                              const float* __restrict__ swb, int* __restrict__ routes,
                              float* __restrict__ pmax_out, float* __restrict__ counts,
                              float* __restrict__ probsum) {
    int tok  = blockIdx.x * 8 + (threadIdx.x >> 5);
    int lane = threadIdx.x & 31;
    const float* tr = t + (size_t)tok * DD;
    float a0 = 0, a1 = 0, a2 = 0, a3 = 0;
    for (int k = lane; k < DD; k += 32) {
        float tv = tr[k];
        const float4 w = *(const float4*)(swW + k * 4);
        a0 += tv * w.x; a1 += tv * w.y; a2 += tv * w.z; a3 += tv * w.w;
    }
    for (int m = 16; m >= 1; m >>= 1) {
        a0 += __shfl_xor(a0, m, 32);
        a1 += __shfl_xor(a1, m, 32);
        a2 += __shfl_xor(a2, m, 32);
        a3 += __shfl_xor(a3, m, 32);
    }
    if (lane == 0) {
        float l[4] = {a0 + swb[0], a1 + swb[1], a2 + swb[2], a3 + swb[3]};
        int best = 0; float bm = l[0];
        for (int e = 1; e < 4; ++e) if (l[e] > bm) { bm = l[e]; best = e; }
        float s = 0.f, p[4];
        for (int e = 0; e < 4; ++e) { p[e] = expf(l[e] - bm); s += p[e]; }
        float inv = 1.0f / s;
        for (int e = 0; e < 4; ++e) atomicAdd(&probsum[e], p[e] * inv);
        atomicAdd(&counts[best], 1.0f);
        routes[tok]   = best;
        pmax_out[tok] = p[best] * inv;
    }
}

// ---------------- Deterministic capacity scan (single block) ----------------
__global__ void scan_kernel(const int* __restrict__ routes, int* __restrict__ slot,
                            float* __restrict__ nd_out) {
    __shared__ int base[4];
    __shared__ int shr[256];
    __shared__ int nd;
    int tid = threadIdx.x;
    if (tid < 4) base[tid] = 0;
    if (tid == 0) nd = 0;
    __syncthreads();
    for (int start = 0; start < T_TOK; start += 256) {
        int r = routes[start + tid];
        shr[tid] = r;
        __syncthreads();
        int pos = base[r];
        for (int j = 0; j < tid; ++j) pos += (shr[j] == r);
        int s;
        if (pos < CAPC) s = r * CAPC + pos;
        else { s = -1; atomicAdd(&nd, 1); }
        slot[start + tid] = s;
        __syncthreads();
        if (tid < 4) {
            int c = 0;
            for (int j = 0; j < 256; ++j) c += (shr[j] == tid);
            base[tid] += c;
        }
        __syncthreads();
    }
    if (tid == 0) nd_out[0] = (float)nd;
}

// ---------------- Dispatch scatter ----------------
__global__ void scatter_kernel(const bf16* __restrict__ tb, const int* __restrict__ slot,
                               bf16* __restrict__ eb) {
    int tok = blockIdx.x;
    int s = slot[tok];
    if (s < 0) return;
    int c = threadIdx.x * 8;
    *(bf16x8*)&eb[(size_t)s * DD + c] = *(const bf16x8*)&tb[(size_t)tok * DD + c];
}

// ---------------- Combine ----------------
__global__ void combine_kernel(const float* __restrict__ yy, const float* __restrict__ tf,
                               const int* __restrict__ slot, const float* __restrict__ pmax,
                               float* __restrict__ out) {
    int tok = blockIdx.x;
    int s = slot[tok];
    float pm = pmax[tok];
    int c = threadIdx.x * 8;
    const float* src = (s >= 0) ? (yy + (size_t)s * DD) : (tf + (size_t)tok * DD);
    float4 v0 = *(const float4*)(src + c);
    float4 v1 = *(const float4*)(src + c + 4);
    float4 o0 = {v0.x * pm, v0.y * pm, v0.z * pm, v0.w * pm};
    float4 o1 = {v1.x * pm, v1.y * pm, v1.z * pm, v1.w * pm};
    *(float4*)(out + (size_t)tok * DD + c)     = o0;
    *(float4*)(out + (size_t)tok * DD + c + 4) = o1;
}

__global__ void zero_stats(float* p) { if (threadIdx.x < 9) p[threadIdx.x] = 0.f; }

extern "C" void kernel_launch(void* const* d_in, const int* in_sizes, int n_in,
                              void* d_out, int out_size, void* d_ws, size_t ws_size,
                              hipStream_t stream) {
    const float* x    = (const float*)d_in[0];
    const float* ln_g = (const float*)d_in[1];
    const float* ln_b = (const float*)d_in[2];
    const float* preW = (const float*)d_in[3];
    const float* preb = (const float*)d_in[4];
    const float* swW  = (const float*)d_in[5];
    const float* swb  = (const float*)d_in[6];
    const float* W1   = (const float*)d_in[7];
    const float* b1   = (const float*)d_in[8];
    const float* W2   = (const float*)d_in[9];
    const float* b2   = (const float*)d_in[10];
    float* out = (float*)d_out;

    char* ws = (char*)d_ws;
    size_t off = 0;
    auto alloc = [&](size_t bytes) -> char* {
        char* p = ws + off;
        off += (bytes + 255) & ~(size_t)255;
        return p;
    };
    bf16*  xn    = (bf16*) alloc((size_t)T_TOK * D_IN * 2);
    bf16*  preWt = (bf16*) alloc((size_t)DD * D_IN * 2);
    float* tf    = (float*)alloc((size_t)T_TOK * DD * 4);
    bf16*  tb    = (bf16*) alloc((size_t)T_TOK * DD * 2);
    bf16*  W1t   = (bf16*) alloc((size_t)EE * HH * DD * 2);
    bf16*  W2t   = (bf16*) alloc((size_t)EE * DD * HH * 2);
    bf16*  eb    = (bf16*) alloc((size_t)SLOTS * DD * 2);
    bf16*  hh    = (bf16*) alloc((size_t)SLOTS * HH * 2);
    float* yy    = (float*)alloc((size_t)SLOTS * DD * 4);
    int*   routes= (int*)  alloc((size_t)T_TOK * 4);
    int*   slot  = (int*)  alloc((size_t)T_TOK * 4);

    float* o_final  = out;
    float* o_counts = out + (size_t)T_TOK * DD;
    float* o_prob   = o_counts + 4;
    float* o_nd     = o_prob + 4;
    float* o_pmax   = o_nd + 1;

    // 1. weight convert + transpose to bf16 [N][K]
    transpose_to_bf16<<<dim3(D_IN / 32, DD / 32, 1),  dim3(32, 8), 0, stream>>>(preW, preWt, D_IN, DD);
    transpose_to_bf16<<<dim3(DD / 32,  HH / 32, EE),  dim3(32, 8), 0, stream>>>(W1, W1t, DD, HH);
    transpose_to_bf16<<<dim3(HH / 32,  DD / 32, EE),  dim3(32, 8), 0, stream>>>(W2, W2t, HH, DD);

    // 2. LayerNorm -> bf16 activations
    ln_to_bf16<<<T_TOK, 256, 0, stream>>>(x, ln_g, ln_b, xn);

    // 3. pre_linear + GELU  (t in f32 for router/passthrough, bf16 for dispatch)
    gemm_wmma_bf16<<<dim3(DD / 128, T_TOK / 128, 1), 256, 0, stream>>>(
        xn, preWt, preb, tf, tb, T_TOK, DD, D_IN, 0, 0, 0, 0, 0, 1);

    // 4. routing
    zero_stats<<<1, 32, 0, stream>>>(o_counts);
    router_kernel<<<T_TOK / 8, 256, 0, stream>>>(tf, swW, swb, routes, o_pmax, o_counts, o_prob);
    scan_kernel<<<1, 256, 0, stream>>>(routes, slot, o_nd);
    scatter_kernel<<<T_TOK, 256, 0, stream>>>(tb, slot, eb);

    // 5. expert FFN (grouped over blockIdx.z)
    gemm_wmma_bf16<<<dim3(HH / 128, CAPC / 128, EE), 256, 0, stream>>>(
        eb, W1t, b1, nullptr, hh, CAPC, HH, DD,
        (long)CAPC * DD, (long)HH * DD, (long)HH, 0, (long)CAPC * HH, 1);

    gemm_wmma_bf16<<<dim3(DD / 128, CAPC / 128, EE), 256, 0, stream>>>(
        hh, W2t, b2, yy, nullptr, CAPC, DD, HH,
        (long)CAPC * HH, (long)DD * HH, (long)DD, (long)CAPC * DD, 0, 0);

    // 6. combine
    combine_kernel<<<T_TOK, 256, 0, stream>>>(yy, tf, slot, o_pmax, o_final);
}
